// GRUCell_764504179296
// MI455X (gfx1250) — compile-verified, matrix operand order checked by reading, not run on hardware
//
#include <hip/hip_runtime.h>
#include <hip/hip_bf16.h>

// ---------------- CDNA5 WMMA types ----------------
typedef __attribute__((ext_vector_type(16))) __bf16 v16bf;
typedef __attribute__((ext_vector_type(8)))  float  v8f;

__device__ __forceinline__ v8f wmma_bf16(v16bf a, v16bf b, v8f c) {
    return __builtin_amdgcn_wmma_f32_16x16x32_bf16(
        /*neg_a=*/false, a, /*neg_b=*/false, b,
        /*c_mod=*/(short)0, c, /*reuse_a=*/false, /*reuse_b=*/false);
}

// Load a 16x32 bf16 fragment from an LDS tile stored row-major with `stride`
// bf16 elements per row. Matches the CDNA5 16-bit A/B operand lane layout:
//   lanes 0-15 : row = row0+lane, halves = [k 0..7], [k 16..23]
//   lanes 16-31: row = row0+lane-16, halves = [k 8..15], [k 24..31]
// Each fragment = two 16-byte LDS loads per lane.
__device__ __forceinline__ v16bf load_frag(const __bf16* base, int row0, int stride, int lane) {
    int r  = row0 + (lane & 15);
    int hi = (lane >> 4) & 1;
    const __bf16* p = base + r * stride + hi * 8;
    union { v16bf v; uint4 q[2]; } u;
    u.q[0] = *(const uint4*)(p);
    u.q[1] = *(const uint4*)(p + 16);
    return u.v;
}

// ---------------- problem constants ----------------
constexpr int B = 64, N = 2048, DIN = 2, H = 64, D = 16;
constexpr int C = DIN + H;        // 66
constexpr int O = H;              // 64
constexpr int J = B * C;          // 4224 flattened (b,c) columns
constexpr int K2C = 2 * C;        // 132
constexpr int KPAD = 160;         // 132 padded to 5*32 for bf16 WMMA

// ---------------- kernel 1: e = layernorm(node_emb + time_emb) ----------------
__global__ void compute_e_kernel(float* __restrict__ e,
                                 const float* __restrict__ nodeE,
                                 const float* __restrict__ timeE,
                                 const float* __restrict__ gamma,
                                 const float* __restrict__ beta) {
    int n = blockIdx.x * blockDim.x + threadIdx.x;
    if (n >= N) return;
    float v[D]; float m = 0.f;
#pragma unroll
    for (int d = 0; d < D; ++d) { v[d] = nodeE[n * D + d] + timeE[d]; m += v[d]; }
    m *= (1.f / D);
    float var = 0.f;
#pragma unroll
    for (int d = 0; d < D; ++d) { float t = v[d] - m; var += t * t; }
    var *= (1.f / D);
    float rs = rsqrtf(var + 1e-12f);
#pragma unroll
    for (int d = 0; d < D; ++d)
        e[n * D + d] = (v[d] - m) * rs * gamma[d] + beta[d];
}

// ---------------- kernel 2: A = softmax_row(e @ e^T), stored bf16 ----------------
__global__ void adjacency_kernel(__bf16* __restrict__ A, const float* __restrict__ e) {
    __shared__ float er[D];
    __shared__ float red[256];
    int n = blockIdx.x, tid = threadIdx.x;
    if (tid < D) er[tid] = e[n * D + tid];
    __syncthreads();
    float en[D];
#pragma unroll
    for (int d = 0; d < D; ++d) en[d] = er[d];

    float s[8];
    float mx = -1e30f;
#pragma unroll
    for (int i = 0; i < 8; ++i) {
        int m = tid + i * 256;
        const float* em = e + m * D;
        float dot = 0.f;
#pragma unroll
        for (int d = 0; d < D; ++d) dot += en[d] * em[d];
        s[i] = dot; mx = fmaxf(mx, dot);
    }
    red[tid] = mx; __syncthreads();
    for (int off = 128; off > 0; off >>= 1) {
        if (tid < off) red[tid] = fmaxf(red[tid], red[tid + off]);
        __syncthreads();
    }
    mx = red[0]; __syncthreads();

    float sum = 0.f;
#pragma unroll
    for (int i = 0; i < 8; ++i) { s[i] = __expf(s[i] - mx); sum += s[i]; }
    red[tid] = sum; __syncthreads();
    for (int off = 128; off > 0; off >>= 1) {
        if (tid < off) red[tid] += red[tid + off];
        __syncthreads();
    }
    float inv = 1.f / red[0];
#pragma unroll
    for (int i = 0; i < 8; ++i)
        A[(size_t)n * N + tid + i * 256] = (__bf16)(s[i] * inv);
}

// ---------------- kernel 3: X'^T[j][m] = concat(x, (z?)*state)[b,m,c], bf16 ----------------
__global__ void build_xsT_kernel(__bf16* __restrict__ XpT,
                                 const float* __restrict__ x,
                                 const float* __restrict__ state,
                                 const float* __restrict__ z) {
    long idx = (long)blockIdx.x * 256 + threadIdx.x;
    if (idx >= (long)J * N) return;
    int m = (int)(idx & (N - 1));
    int j = (int)(idx >> 11);
    int b = j / C, c = j - b * C;
    size_t bm = (size_t)b * N + m;
    float val;
    if (c < DIN) {
        val = x[bm * DIN + c];
    } else {
        float s = state[bm * H + (c - DIN)];
        if (z) s *= z[bm * H + (c - DIN)];
        val = s;
    }
    XpT[(size_t)j * N + m] = (__bf16)val;
}

// ---------------- kernel 4: Y[n][j] = sum_m A[n][m] * X'^T[j][m]  (bf16 WMMA) ----------------
// Block = 256 threads = 8 waves; block tile 128 rows(n) x 64 cols(j); K step 32.
__global__ void gemm_kernel(__bf16* __restrict__ Y,
                            const __bf16* __restrict__ A,
                            const __bf16* __restrict__ XpT) {
    __shared__ __align__(16) __bf16 Atile[128 * 32];
    __shared__ __align__(16) __bf16 Btile[64 * 32];
    int tid = threadIdx.x;
    int n0 = blockIdx.y * 128;
    int j0 = blockIdx.x * 64;
    int w = tid >> 5, lane = tid & 31;
    int wm = w & 3;        // 4 row groups of 32
    int wn = w >> 2;       // 2 col groups of 32

    v8f acc[2][2] = {{{}, {}}, {{}, {}}};

    for (int kb = 0; kb < N / 32; ++kb) {
        int k0 = kb * 32;
        __syncthreads();
        // stage A tile: 128 rows x 32 bf16 = 512 uint4 chunks (2 passes of 256)
#pragma unroll
        for (int p = 0; p < 2; ++p) {
            int lin = p * 256 + tid;
            int row = lin >> 2, chunk = lin & 3;
            ((uint4*)Atile)[lin] =
                *(const uint4*)(A + (size_t)(n0 + row) * N + k0 + chunk * 8);
        }
        // stage B tile (stored as [j][k]): 64 rows x 32 bf16 = 256 uint4 chunks
        {
            int row = tid >> 2, chunk = tid & 3;
            ((uint4*)Btile)[tid] =
                *(const uint4*)(XpT + (size_t)(j0 + row) * N + k0 + chunk * 8);
        }
        __syncthreads();

        v16bf a0 = load_frag(Atile, wm * 32, 32, lane);
        v16bf a1 = load_frag(Atile, wm * 32 + 16, 32, lane);
        v16bf b0 = load_frag(Btile, wn * 32, 32, lane);
        v16bf b1 = load_frag(Btile, wn * 32 + 16, 32, lane);

        acc[0][0] = wmma_bf16(a0, b0, acc[0][0]);
        acc[0][1] = wmma_bf16(a0, b1, acc[0][1]);
        acc[1][0] = wmma_bf16(a1, b0, acc[1][0]);
        acc[1][1] = wmma_bf16(a1, b1, acc[1][1]);
    }

    int nc = lane & 15, hi = lane >> 4;
#pragma unroll
    for (int sm = 0; sm < 2; ++sm)
#pragma unroll
        for (int sn = 0; sn < 2; ++sn) {
            v8f a = acc[sm][sn];
            int rowb = n0 + wm * 32 + sm * 16 + hi * 8;
            int col  = j0 + wn * 32 + sn * 16 + nc;
#pragma unroll
            for (int v = 0; v < 8; ++v)
                Y[(size_t)(rowb + v) * J + col] = (__bf16)a[v];
        }
}

// ---------------- kernel 5: per-node adaptive GEMM + activation ----------------
// One block per node n. Wn = e[n] . W_pool built in LDS (bf16, [o][k] layout),
// Act = [batch][k] with k = concat(xs, A@xs) padded 132->160.
// 8 waves: wave tile 16(b) x 32(o), 5 K-steps -> 10 WMMAs/wave.
__global__ void gate_out_kernel(float* __restrict__ out,
                                const float* __restrict__ e,
                                const float* __restrict__ W,     // [16][8448]
                                const float* __restrict__ bp,    // [16][64]
                                const float* __restrict__ x,
                                const float* __restrict__ state,
                                const float* __restrict__ zgate, // null for z/r
                                const __bf16* __restrict__ Y,
                                const float* __restrict__ rgate) // null => sigmoid, else final
{
    __shared__ __align__(16) __bf16 Act[B * KPAD];
    __shared__ __align__(16) __bf16 Wt[O * KPAD];
    __shared__ float en[D];
    __shared__ float bias[O];

    int n = blockIdx.x, tid = threadIdx.x;
    if (tid < D) en[tid] = e[n * D + tid];
    __syncthreads();
    float el[D];
#pragma unroll
    for (int d = 0; d < D; ++d) el[d] = en[d];

    // node-adaptive weights -> Wt[o][k]
    for (int idx = tid; idx < O * KPAD; idx += 256) {
        int ki = idx >> 6, o = idx & 63;
        float acc = 0.f;
        if (ki < K2C) {
            const float* wp = W + (size_t)ki * O + o;
#pragma unroll
            for (int d = 0; d < D; ++d) acc += el[d] * wp[(size_t)d * (K2C * O)];
        }
        Wt[o * KPAD + ki] = (__bf16)acc;
    }
    if (tid < O) {
        float acc = 0.f;
#pragma unroll
        for (int d = 0; d < D; ++d) acc += el[d] * bp[d * O + tid];
        bias[tid] = acc;
    }
    // activations -> Act[b][k]
    for (int idx = tid; idx < B * KPAD; idx += 256) {
        int b = idx / KPAD, ki = idx - b * KPAD;
        size_t bn = (size_t)b * N + n;
        float val = 0.f;
        if (ki < DIN) {
            val = x[bn * DIN + ki];
        } else if (ki < C) {
            float s = state[bn * H + (ki - DIN)];
            if (zgate) s *= zgate[bn * H + (ki - DIN)];
            val = s;
        } else if (ki < K2C) {
            val = (float)Y[(size_t)n * J + (size_t)b * C + (ki - C)];
        }
        Act[b * KPAD + ki] = (__bf16)val;
    }
    __syncthreads();

    int w = tid >> 5, lane = tid & 31;
    int b0 = (w & 3) * 16;
    int o0 = (w >> 2) * 32;
    v8f acc0 = {}, acc1 = {};
#pragma unroll
    for (int kk = 0; kk < KPAD / 32; ++kk) {
        v16bf a  = load_frag(Act + kk * 32, b0, KPAD, lane);
        v16bf w0 = load_frag(Wt + kk * 32, o0, KPAD, lane);
        v16bf w1 = load_frag(Wt + kk * 32, o0 + 16, KPAD, lane);
        acc0 = wmma_bf16(a, w0, acc0);
        acc1 = wmma_bf16(a, w1, acc1);
    }

    int nc = lane & 15, hi = lane >> 4;
#pragma unroll
    for (int t = 0; t < 2; ++t) {
        v8f a = t ? acc1 : acc0;
        int o = o0 + t * 16 + nc;
#pragma unroll
        for (int v = 0; v < 8; ++v) {
            int b = b0 + hi * 8 + v;
            float val = a[v] + bias[o];
            size_t idx = ((size_t)b * N + n) * (size_t)O + o;
            if (rgate) {
                float r = rgate[idx];
                out[idx] = r * state[idx] + (1.f - r) * tanhf(val);
            } else {
                out[idx] = 1.f / (1.f + __expf(-val));
            }
        }
    }
}

// ---------------- host side ----------------
extern "C" void kernel_launch(void* const* d_in, const int* in_sizes, int n_in,
                              void* d_out, int out_size, void* d_ws, size_t ws_size,
                              hipStream_t stream) {
    const float* x      = (const float*)d_in[0];
    const float* state  = (const float*)d_in[1];
    const float* nodeE  = (const float*)d_in[2];
    const float* timeE  = (const float*)d_in[3];
    const float* Wg[3]  = {(const float*)d_in[4],  (const float*)d_in[8],  (const float*)d_in[12]};
    const float* bg[3]  = {(const float*)d_in[5],  (const float*)d_in[9],  (const float*)d_in[13]};
    const float* gam[3] = {(const float*)d_in[6],  (const float*)d_in[10], (const float*)d_in[14]};
    const float* bet[3] = {(const float*)d_in[7],  (const float*)d_in[11], (const float*)d_in[15]};

    char* ws = (char*)d_ws;
    constexpr size_t OFF_E   = 0;                                       // 2048*16*4
    constexpr size_t OFF_A   = OFF_E   + (size_t)N * D * 4;             // 2048*2048*2
    constexpr size_t OFF_XPT = OFF_A   + (size_t)N * N * 2;             // 4224*2048*2
    constexpr size_t OFF_Y   = OFF_XPT + (size_t)J * N * 2;             // 2048*4224*2
    constexpr size_t OFF_Z   = OFF_Y   + (size_t)N * J * 2;             // B*N*H*4
    constexpr size_t OFF_R   = OFF_Z   + (size_t)B * N * H * 4;

    float*  e_ws  = (float*) (ws + OFF_E);
    __bf16* A_ws  = (__bf16*)(ws + OFF_A);
    __bf16* Xp_ws = (__bf16*)(ws + OFF_XPT);
    __bf16* Y_ws  = (__bf16*)(ws + OFF_Y);
    float*  z_ws  = (float*) (ws + OFF_Z);
    float*  r_ws  = (float*) (ws + OFF_R);
    float*  outp  = (float*)d_out;

    const dim3 gemm_grid(J / 64, N / 128);           // 66 x 16
    const unsigned xsT_blocks = ((unsigned)((long)J * N + 255)) / 256;  // 33792

    // ---- gate z ----
    compute_e_kernel<<<N / 256, 256, 0, stream>>>(e_ws, nodeE, timeE, gam[0], bet[0]);
    adjacency_kernel<<<N, 256, 0, stream>>>(A_ws, e_ws);
    build_xsT_kernel<<<xsT_blocks, 256, 0, stream>>>(Xp_ws, x, state, nullptr);
    gemm_kernel<<<gemm_grid, 256, 0, stream>>>(Y_ws, A_ws, Xp_ws);
    gate_out_kernel<<<N, 256, 0, stream>>>(z_ws, e_ws, Wg[0], bg[0], x, state,
                                           nullptr, Y_ws, nullptr);
    // ---- gate r (reuses X'^T from z) ----
    compute_e_kernel<<<N / 256, 256, 0, stream>>>(e_ws, nodeE, timeE, gam[1], bet[1]);
    adjacency_kernel<<<N, 256, 0, stream>>>(A_ws, e_ws);
    gemm_kernel<<<gemm_grid, 256, 0, stream>>>(Y_ws, A_ws, Xp_ws);
    gate_out_kernel<<<N, 256, 0, stream>>>(r_ws, e_ws, Wg[1], bg[1], x, state,
                                           nullptr, Y_ws, nullptr);
    // ---- gate u + fused final combine ----
    compute_e_kernel<<<N / 256, 256, 0, stream>>>(e_ws, nodeE, timeE, gam[2], bet[2]);
    adjacency_kernel<<<N, 256, 0, stream>>>(A_ws, e_ws);
    build_xsT_kernel<<<xsT_blocks, 256, 0, stream>>>(Xp_ws, x, state, z_ws);
    gemm_kernel<<<gemm_grid, 256, 0, stream>>>(Y_ws, A_ws, Xp_ws);
    gate_out_kernel<<<N, 256, 0, stream>>>(outp, e_ws, Wg[2], bg[2], x, state,
                                           z_ws, Y_ws, r_ws);
}
